// Attention_49770081026258
// MI455X (gfx1250) — matrix operand order repaired and checked by reading, not compiled, not run on hardware
//
#include <hip/hip_runtime.h>
#include <hip/hip_bf16.h>

typedef __attribute__((ext_vector_type(16))) __bf16 v16bf;
typedef __attribute__((ext_vector_type(8)))  __bf16 v8bf;
typedef __attribute__((ext_vector_type(8)))  float  v8f;
typedef __attribute__((ext_vector_type(4)))  int    v4i;

#define AS1 __attribute__((address_space(1)))
#define AS3 __attribute__((address_space(3)))

#if defined(__has_builtin)
#if __has_builtin(__builtin_amdgcn_global_load_async_to_lds_b128)
#define USE_ASYNC_LDS 1
#endif
#endif

// ---------------------------------------------------------------------------
// 16-byte global -> LDS copy: async (ASYNCcnt-tracked DMA) when available,
// else synchronous vector copy through VGPRs.
// ---------------------------------------------------------------------------
__device__ inline void copy16_g2l(const __bf16* g, __bf16* l) {
#ifdef USE_ASYNC_LDS
  __builtin_amdgcn_global_load_async_to_lds_b128(
      (AS1 v4i*)(void*)const_cast<__bf16*>(g),
      (AS3 v4i*)(void*)l, /*offset=*/0, /*cpol=*/0);
#else
  *reinterpret_cast<v8bf*>(l) = *reinterpret_cast<const v8bf*>(g);
#endif
}

__device__ inline void wait_async_copies() {
#ifdef USE_ASYNC_LDS
#if __has_builtin(__builtin_amdgcn_s_wait_asynccnt)
  __builtin_amdgcn_s_wait_asynccnt(0);
#else
  asm volatile("s_wait_asynccnt 0" ::: "memory");
#endif
#endif
}

// ---------------------------------------------------------------------------
// WMMA wrapper: D = A(16x32 bf16) x B(32x16 bf16) + C(16x16 f32)
// ---------------------------------------------------------------------------
__device__ inline v8f wmma_bf16(v16bf a, v16bf b, v8f c) {
  return __builtin_amdgcn_wmma_f32_16x16x32_bf16(
      /*neg_a=*/false, a, /*neg_b=*/false, b,
      /*c_mod=*/(short)0, c, /*reuse_a=*/false, /*reuse_b=*/false);
}

// A-matrix fragment (16x32, bf16). Per ISA 7.12.2:
//   lanes 0-15 : row M=lane,    elements 0-7 <- K=0..7,  8-15 <- K=16..23
//   lanes16-31 : row M=lane-16, elements 0-7 <- K=8..15, 8-15 <- K=24..31
__device__ inline v16bf load_a_frag(const __bf16* base, int ldk, int lane) {
  const int r = lane & 15, hi = lane >> 4;
  const __bf16* row = base + (size_t)r * ldk;
  v8bf x0 = *reinterpret_cast<const v8bf*>(row + hi * 8);
  v8bf x1 = *reinterpret_cast<const v8bf*>(row + 16 + hi * 8);
  v16bf a;
#pragma unroll
  for (int i = 0; i < 8; ++i) { a[i] = x0[i]; a[i + 8] = x1[i]; }
  return a;
}

// B-matrix fragment (32x16, bf16) read from an n-major (k-contiguous) region:
//   lanes 0-15 : col N=lane,    elements 0-7 <- K=0..7,  8-15 <- K=16..23
//   lanes16-31 : col N=lane-16, elements 0-7 <- K=8..15, 8-15 <- K=24..31
__device__ inline v16bf load_b_frag(const __bf16* base, int ldk, int lane) {
  const int r = lane & 15, hi = lane >> 4;
  const __bf16* row = base + (size_t)r * ldk + hi * 8;
  v8bf x0 = *reinterpret_cast<const v8bf*>(row);
  v8bf x1 = *reinterpret_cast<const v8bf*>(row + 16);
  v16bf b;
#pragma unroll
  for (int i = 0; i < 8; ++i) { b[i] = x0[i]; b[i + 8] = x1[i]; }
  return b;
}

// ---------------------------------------------------------------------------
// Elementwise f32 -> bf16 convert
// ---------------------------------------------------------------------------
__global__ __launch_bounds__(256) void cvt_f32_bf16(const float* __restrict__ in,
                                                    __bf16* __restrict__ out,
                                                    long long n) {
  long long i = (long long)blockIdx.x * blockDim.x + threadIdx.x;
  if (i < n) out[i] = (__bf16)in[i];
}

// ---------------------------------------------------------------------------
// NT GEMM: C(MxN) = A(MxK, row-major) @ B(NxK, row-major)^T  [+ bias]
// bf16 inputs, f32 accumulate, bf16 or f32 output. Async-copy staged tiles.
// ---------------------------------------------------------------------------
template <int BM, int BN, bool F32OUT, bool BIAS>
__global__ __launch_bounds__(256) void gemm_bf16_nt(
    const __bf16* __restrict__ A, int lda, const __bf16* __restrict__ Bm, int ldb,
    void* __restrict__ Cout, int ldc, int M, int N, int K,
    const float* __restrict__ bias) {
  constexpr int WX = (BN >= 128) ? 2 : 1;
  constexpr int WY = 8 / WX;
  constexpr int WM = BM / WY;
  constexpr int WN = BN / WX;
  constexpr int MT = WM / 16;
  constexpr int NT = WN / 16;

  __shared__ __bf16 sA[BM][32];
  __shared__ __bf16 sB[BN][32];

  const int tid = threadIdx.x;
  const int lane = tid & 31;
  const int w = tid >> 5;
  const int wy = w % WY, wx = w / WY;
  const int m0 = blockIdx.y * BM;
  const int n0 = blockIdx.x * BN;

  v8f acc[MT][NT];
#pragma unroll
  for (int i = 0; i < MT; ++i)
#pragma unroll
    for (int j = 0; j < NT; ++j)
#pragma unroll
      for (int e = 0; e < 8; ++e) acc[i][j][e] = 0.0f;

  for (int k0 = 0; k0 < K; k0 += 32) {
    for (int i = tid; i < BM * 4; i += 256) {
      int row = i >> 2, c = (i & 3) * 8;
      copy16_g2l(A + (size_t)(m0 + row) * lda + k0 + c, &sA[row][c]);
    }
    for (int i = tid; i < BN * 4; i += 256) {
      int row = i >> 2, c = (i & 3) * 8;
      copy16_g2l(Bm + (size_t)(n0 + row) * ldb + k0 + c, &sB[row][c]);
    }
    if (k0 + 32 < K) {  // prefetch next K-tile (one line per thread)
      __builtin_prefetch(A + (size_t)(m0 + (tid >> 1)) * lda + k0 + 32, 0, 0);
      __builtin_prefetch(Bm + (size_t)(n0 + (tid & 127)) * ldb + k0 + 32, 0, 0);
    }
    wait_async_copies();
    __syncthreads();

    v16bf af[MT], bfv[NT];
#pragma unroll
    for (int i = 0; i < MT; ++i) af[i] = load_a_frag(&sA[wy * WM + i * 16][0], 32, lane);
#pragma unroll
    for (int j = 0; j < NT; ++j) bfv[j] = load_b_frag(&sB[wx * WN + j * 16][0], 32, lane);
#pragma unroll
    for (int i = 0; i < MT; ++i)
#pragma unroll
      for (int j = 0; j < NT; ++j) acc[i][j] = wmma_bf16(af[i], bfv[j], acc[i][j]);
    __syncthreads();
  }

  const int r = lane & 15, hi = lane >> 4;
#pragma unroll
  for (int i = 0; i < MT; ++i)
#pragma unroll
    for (int j = 0; j < NT; ++j) {
      int row = m0 + wy * WM + i * 16 + hi * 8;
      int col = n0 + wx * WN + j * 16 + r;
      float bv = BIAS ? bias[col] : 0.0f;
#pragma unroll
      for (int v = 0; v < 8; ++v) {
        float val = acc[i][j][v] + bv;
        if (F32OUT)
          reinterpret_cast<float*>(Cout)[(size_t)(row + v) * ldc + col] = val;
        else
          reinterpret_cast<__bf16*>(Cout)[(size_t)(row + v) * ldc + col] = (__bf16)val;
      }
    }
}

// ---------------------------------------------------------------------------
// NN GEMM: C(MxN) = A0(MxK)@B0(KxN) [+ A1(MxK)@B1(KxN) if DUAL], bf16 out.
// HEADS variant applies per-(b,h) base offsets from blockIdx.z.
// A staged with async copies; B transpose-staged (n-major) in LDS.
// ---------------------------------------------------------------------------
template <int BM, int BN, bool DUAL, bool HEADS>
__global__ __launch_bounds__(256) void gemm_bf16_nn(
    const __bf16* __restrict__ A0, int lda, const __bf16* __restrict__ B0, int ldb,
    const __bf16* __restrict__ A1, const __bf16* __restrict__ B1,
    __bf16* __restrict__ Cout, int ldc, int M, int N, int K) {
  constexpr int WX = (BN >= 128) ? 2 : 1;
  constexpr int WY = 8 / WX;
  constexpr int WM = BM / WY;
  constexpr int WN = BN / WX;
  constexpr int MT = WM / 16;
  constexpr int NT = WN / 16;
  constexpr int CH = BN / 8;

  __shared__ __bf16 sA[BM][32];
  __shared__ __bf16 sB[BN][32];  // stored n-major: sB[n][k]

  const int tid = threadIdx.x;
  const int lane = tid & 31;
  const int w = tid >> 5;
  const int wy = w % WY, wx = w / WY;
  const int m0 = blockIdx.y * BM;
  const int n0 = blockIdx.x * BN;

  const __bf16* Ap0 = A0;
  const __bf16* Bp0 = B0;
  __bf16* Cp = Cout;
  if (HEADS) {
    int bh = blockIdx.z;
    int b = bh / 12, h = bh % 12;
    Ap0 += (size_t)bh << 20;  // 1024*1024 per (b,h)
    Bp0 += ((size_t)b * 1024) * 768 + (size_t)h * 64;
    Cp += ((size_t)b * 1024) * 768 + (size_t)h * 64;
  }
  const __bf16* As[2] = {Ap0, A1};
  const __bf16* Bs[2] = {Bp0, B1};

  v8f acc[MT][NT];
#pragma unroll
  for (int i = 0; i < MT; ++i)
#pragma unroll
    for (int j = 0; j < NT; ++j)
#pragma unroll
      for (int e = 0; e < 8; ++e) acc[i][j][e] = 0.0f;

  const int npairs = DUAL ? 2 : 1;
  for (int pp = 0; pp < npairs; ++pp) {
    const __bf16* Ap = As[pp];
    const __bf16* Bp = Bs[pp];
    for (int k0 = 0; k0 < K; k0 += 32) {
      for (int i = tid; i < BM * 4; i += 256) {
        int row = i >> 2, c = (i & 3) * 8;
        copy16_g2l(Ap + (size_t)(m0 + row) * lda + k0 + c, &sA[row][c]);
      }
      // transpose-stage B: global B[k][n] (n contiguous) -> sB[n][k]
      for (int i = tid; i < 32 * CH; i += 256) {
        int k = i / CH, nc = (i % CH) * 8;
        v8bf vv = *reinterpret_cast<const v8bf*>(Bp + (size_t)(k0 + k) * ldb + n0 + nc);
#pragma unroll
        for (int j = 0; j < 8; ++j) sB[nc + j][k] = vv[j];
      }
      wait_async_copies();
      __syncthreads();

      v16bf af[MT], bfv[NT];
#pragma unroll
      for (int i = 0; i < MT; ++i) af[i] = load_a_frag(&sA[wy * WM + i * 16][0], 32, lane);
#pragma unroll
      for (int j = 0; j < NT; ++j) bfv[j] = load_b_frag(&sB[wx * WN + j * 16][0], 32, lane);
#pragma unroll
      for (int i = 0; i < MT; ++i)
#pragma unroll
        for (int j = 0; j < NT; ++j) acc[i][j] = wmma_bf16(af[i], bfv[j], acc[i][j]);
      __syncthreads();
    }
  }

  const int r = lane & 15, hi = lane >> 4;
#pragma unroll
  for (int i = 0; i < MT; ++i)
#pragma unroll
    for (int j = 0; j < NT; ++j) {
      int row = m0 + wy * WM + i * 16 + hi * 8;
      int col = n0 + wx * WN + j * 16 + r;
#pragma unroll
      for (int v = 0; v < 8; ++v)
        Cp[(size_t)(row + v) * ldc + col] = (__bf16)acc[i][j][v];
    }
}

// ---------------------------------------------------------------------------
// Fused attention scores + softmax, writing A^T.
// One block = one (b, h, 16-row q strip). 8 waves each compute a 16x128 slab
// of S = q_h @ k_h^T * scale via WMMA with fragments loaded straight from
// global (z contiguous -> pure b128 loads). All 8 B-fragments are issued
// before the WMMA burst so loads overlap the matrix pipe.
// ---------------------------------------------------------------------------
__global__ __launch_bounds__(256) void attn_score_softmax(
    const __bf16* __restrict__ qb, const __bf16* __restrict__ kb,
    __bf16* __restrict__ At, float scale) {
  const int strip = blockIdx.x;  // 0..63
  const int h = blockIdx.y;      // 0..11
  const int b = blockIdx.z;      // 0..7
  const int q0 = strip * 16;
  const int tid = threadIdx.x;
  const int lane = tid & 31;
  const int w = tid >> 5;

  __shared__ __bf16 sS[16][1024];
  __shared__ float red[16][16];
  __shared__ float rmax[16];
  __shared__ float rsum[16];

  const __bf16* qh = qb + ((size_t)b * 1024 + q0) * 768 + (size_t)h * 64;
  const __bf16* kh = kb + ((size_t)b * 1024) * 768 + (size_t)h * 64;

  v8f acc[8];
#pragma unroll
  for (int t = 0; t < 8; ++t)
#pragma unroll
    for (int e = 0; e < 8; ++e) acc[t][e] = 0.0f;

#pragma unroll
  for (int kit = 0; kit < 2; ++kit) {
    const int z0 = kit * 32;
    v16bf a = load_a_frag(qh + z0, 768, lane);
    v16bf bbv[8];
#pragma unroll
    for (int t = 0; t < 8; ++t) {
      const int n0 = w * 128 + t * 16;
      bbv[t] = load_b_frag(kh + (size_t)n0 * 768 + z0, 768, lane);
    }
#pragma unroll
    for (int t = 0; t < 8; ++t) acc[t] = wmma_bf16(a, bbv[t], acc[t]);
  }

  const int r = lane & 15, hi = lane >> 4;
#pragma unroll
  for (int t = 0; t < 8; ++t)
#pragma unroll
    for (int v = 0; v < 8; ++v)
      sS[hi * 8 + v][w * 128 + t * 16 + r] = (__bf16)(acc[t][v] * scale);
  __syncthreads();

  const int row = tid >> 4, col16 = tid & 15;  // 16 threads per row
  float m = -1e30f;
  for (int j = 0; j < 64; ++j) m = fmaxf(m, (float)sS[row][col16 + j * 16]);
  red[row][col16] = m;
  __syncthreads();
  if (col16 == 0) {
    float mm = red[row][0];
    for (int j = 1; j < 16; ++j) mm = fmaxf(mm, red[row][j]);
    rmax[row] = mm;
  }
  __syncthreads();

  const float mrow = rmax[row];
  float s = 0.0f;
  for (int j = 0; j < 64; ++j) {
    int c = col16 + j * 16;
    float p = __expf((float)sS[row][c] - mrow);
    sS[row][c] = (__bf16)p;
    s += p;
  }
  red[row][col16] = s;
  __syncthreads();
  if (col16 == 0) {
    float ss = 0.0f;
    for (int j = 0; j < 16; ++j) ss += red[row][j];
    rsum[row] = ss;
  }
  __syncthreads();

  const float inv = 1.0f / rsum[row];
  __bf16* Atp = At + ((size_t)(b * 12 + h) << 20);  // 1024*1024 per (b,h)
  for (int j = 0; j < 64; ++j) {
    int c = col16 + j * 16;
    Atp[(size_t)c * 1024 + (q0 + row)] = (__bf16)((float)sS[row][c] * inv);
  }
}

// ---------------------------------------------------------------------------
// Host launch
// ---------------------------------------------------------------------------
extern "C" void kernel_launch(void* const* d_in, const int* in_sizes, int n_in,
                              void* d_out, int out_size, void* d_ws, size_t ws_size,
                              hipStream_t stream) {
  const float* x = (const float*)d_in[0];
  const float* Wq = (const float*)d_in[1];
  const float* Wk = (const float*)d_in[2];
  const float* Wp = (const float*)d_in[3];
  const float* bp = (const float*)d_in[4];

  const int B = 8, N = 1024, C = 768, H = 12;
  const long long xN = (long long)B * N * C;  // 6,291,456
  const long long wN = (long long)C * C;      // 589,824
  const float scale = 0.125f;                 // 64^-0.5

  char* p = (char*)d_ws;
  auto carve = [&](size_t bytes) -> void* {
    void* r = (void*)p;
    p += (bytes + 255) & ~(size_t)255;
    return r;
  };
  __bf16* xb = (__bf16*)carve((size_t)xN * 2);
  __bf16* Wqb = (__bf16*)carve((size_t)wN * 2);
  __bf16* Wkb = (__bf16*)carve((size_t)wN * 2);
  __bf16* Wpb = (__bf16*)carve((size_t)wN * 2);
  __bf16* qb = (__bf16*)carve((size_t)xN * 2);
  __bf16* kb = (__bf16*)carve((size_t)xN * 2);
  __bf16* At = (__bf16*)carve((size_t)B * H * N * N * 2);  // 192 MB
  __bf16* T1 = (__bf16*)carve((size_t)xN * 2);
  __bf16* T2 = (__bf16*)carve((size_t)xN * 2);
  __bf16* midb = (__bf16*)carve((size_t)xN * 2);

  // 0) f32 -> bf16 conversions
  cvt_f32_bf16<<<dim3((unsigned)((xN + 255) / 256)), 256, 0, stream>>>(x, xb, xN);
  cvt_f32_bf16<<<dim3((unsigned)((wN + 255) / 256)), 256, 0, stream>>>(Wq, Wqb, wN);
  cvt_f32_bf16<<<dim3((unsigned)((wN + 255) / 256)), 256, 0, stream>>>(Wk, Wkb, wN);
  cvt_f32_bf16<<<dim3((unsigned)((wN + 255) / 256)), 256, 0, stream>>>(Wp, Wpb, wN);

  // 1) q = xb @ Wq^T ; k = xb @ Wk^T   (M=8192, N=768, K=768)
  gemm_bf16_nt<128, 128, false, false>
      <<<dim3(C / 128, (B * N) / 128), 256, 0, stream>>>(
          xb, C, Wqb, C, (void*)qb, C, B * N, C, C, nullptr);
  gemm_bf16_nt<128, 128, false, false>
      <<<dim3(C / 128, (B * N) / 128), 256, 0, stream>>>(
          xb, C, Wkb, C, (void*)kb, C, B * N, C, C, nullptr);

  // 2) A^T = softmax(q k^T * scale)^T per (b,h)
  attn_score_softmax<<<dim3(N / 16, H, B), 256, 0, stream>>>(qb, kb, At, scale);

  // 3) T1 = A^T @ k_h, T2 = A^T @ q_h  (per (b,h): M=1024, N=64, K=1024)
  gemm_bf16_nn<128, 64, false, true>
      <<<dim3(1, N / 128, B * H), 256, 0, stream>>>(
          At, 1024, kb, C, nullptr, nullptr, T1, C, N, 64, N);
  gemm_bf16_nn<128, 64, false, true>
      <<<dim3(1, N / 128, B * H), 256, 0, stream>>>(
          At, 1024, qb, C, nullptr, nullptr, T2, C, N, 64, N);

  // 4) mid = T1 @ Wq + T2 @ Wk  (M=8192, N=768, K=768, dual pair)
  gemm_bf16_nn<128, 128, true, false>
      <<<dim3(C / 128, (B * N) / 128), 256, 0, stream>>>(
          T1, C, Wqb, C, T2, Wkb, midb, C, B * N, C, C);

  // 5) out = mid @ Wp^T + bp  (f32 output)
  gemm_bf16_nt<128, 128, true, true>
      <<<dim3(C / 128, (B * N) / 128), 256, 0, stream>>>(
          midb, C, Wpb, C, d_out, C, B * N, C, C, bp);
}
